// GraphMessagePassing_171798692411
// MI455X (gfx1250) — matrix operand order repaired and checked by reading, not compiled, not run on hardware
//
#include <hip/hip_runtime.h>

#define B_ 2
#define N_ 1000
#define E_ 2000
#define H_ 128

typedef __attribute__((ext_vector_type(2)))  float  v2f;
typedef __attribute__((ext_vector_type(8)))  float  v8f;
typedef __attribute__((ext_vector_type(16))) __bf16 v16bf;

// V_WMMA_F32_16X16X4_F32: exact f32, K=4
__device__ __forceinline__ v8f wmma4(v2f a, v2f b, v8f c) {
  return __builtin_amdgcn_wmma_f32_16x16x4_f32(false, a, false, b, (short)0, c, false, false);
}
// V_WMMA_F32_16X16X32_BF16: K=32, f32 accumulate
__device__ __forceinline__ v8f wmma_bf16(v16bf a, v16bf b, v8f c) {
  return __builtin_amdgcn_wmma_f32_16x16x32_bf16(false, a, false, b, (short)0, c, false, false);
}

union Frag { v16bf v; uint4 q[2]; };

// ---------------------------------------------------------------------------
// Prep: split EN (re-laid-out as B[k=(h*128+j)][i]) into bf16 hi/lo, stored in
// this code's contiguous-K 32x16 B-fragment order (A frags use the same):
//   lane: column i = t*16 + lane%16 ; K-half = (lane<16 ? 0 : 16)
//   v16bf element e -> K = kb*32 + K-half + e   (contiguous 16 K per lane)
// Flat slot = ((t*512 + kb)*32 + lane), 16 bf16 per slot (32B, coalesced).
// hi = truncate-top-16(f32); lo = truncate-top-16(f - hi)  (error ~2^-17 rel)
// ---------------------------------------------------------------------------
__global__ void en_split_kernel(const float* __restrict__ en,
                                unsigned short* __restrict__ hi,
                                unsigned short* __restrict__ lo) {
  int slot = blockIdx.x * blockDim.x + threadIdx.x;  // 8*512*32 = 131072
  int lane = slot & 31;
  int kb   = (slot >> 5) & 511;
  int t    = slot >> 14;
  int i    = (t << 4) + (lane & 15);
  int kbase = (kb << 5) + ((lane & 16) ? 16 : 0);    // multiple of 16
  int h = kbase >> 7, j0 = kbase & 127;              // 16 K stay within one h
  const float* src = en + ((size_t)h * H_ + i) * H_ + j0;
  size_t o = (size_t)slot << 4;
#pragma unroll
  for (int e = 0; e < 16; ++e) {
    float f = src[e];
    unsigned int b  = __float_as_uint(f);
    unsigned int hb = b & 0xFFFF0000u;
    float rem = f - __uint_as_float(hb);
    hi[o + e] = (unsigned short)(b >> 16);
    lo[o + e] = (unsigned short)(__float_as_uint(rem) >> 16);
  }
}

// -------- norm[b,n] = 1 + sum_e edge2node[b,n,e] ---------------------------
__global__ void norm_kernel(const float* __restrict__ e2n,
                            float* __restrict__ norm) {
  int row = blockIdx.x;  // 0 .. B*N-1
  const float* p = e2n + (size_t)row * E_;
  float s = 0.f;
  for (int e = threadIdx.x; e < E_; e += 256) s += p[e];
  __shared__ float red[256];
  red[threadIdx.x] = s;
  __syncthreads();
  for (int off = 128; off > 0; off >>= 1) {
    if (threadIdx.x < off) red[threadIdx.x] += red[threadIdx.x + off];
    __syncthreads();
  }
  if (threadIdx.x == 0) norm[row] = red[0] + 1.0f;
}

// -------- GEMM1: emb[b,e,h] = sum_n node2edge[b,e,n]*node_state[b,n,h] -----
// exact f32 WMMA; one wave per 16x16 tile; M=2000, K=1000, N=128
__global__ void gemm1_kernel(const float* __restrict__ n2e,
                             const float* __restrict__ ns,
                             float* __restrict__ emb) {
  int lane = threadIdx.x & 31;
  int m = lane & 15;
  int koff = (lane >> 4) << 1;
  int e0 = blockIdx.x << 4;
  int i0 = blockIdx.y << 4;
  int b = blockIdx.z;
  const float* A  = n2e + ((size_t)b * E_ + e0 + m) * N_ + koff;
  const float* Bp = ns + (size_t)b * N_ * H_ + (size_t)koff * H_ + i0 + m;
  v8f acc = {};
#pragma unroll 4
  for (int k = 0; k < N_; k += 4) {
    float2 av = *(const float2*)(A + k);
    v2f a = {av.x, av.y};
    v2f bb = {Bp[(size_t)k * H_], Bp[(size_t)(k + 1) * H_]};
    acc = wmma4(a, bb, acc);
  }
  float* C = emb + ((size_t)b * E_ + e0 + ((lane >> 4) << 3)) * H_ + i0 + m;
#pragma unroll
  for (int r = 0; r < 8; ++r) C[(size_t)r * H_] = acc[r];
}

// ---------------------------------------------------------------------------
// GEMM2 (dominant 16.8 GFLOP): new_n2e[b,e,i] = sum_{h,j} ev*emb*EN
// Outer-product GEMM, M=16 rows/block, K=16384, N=128 (8 waves = 8 i-tiles).
// Split-bf16 3-product (Ah*Bh + Ah*Bl + Al*Bh), f32 accumulate.
// A fragments (split of P = ev x emb) produced ONCE per block into LDS in
// contiguous-K A-fragment order (as B), shared by all 8 waves (8x VALU reduction):
//   A 16-bit 16x32 layout: lane row m = lane%16,
//   element e -> K = kb*32 + (lane<16?0:16) + e  (contiguous 16 K per lane)
// ---------------------------------------------------------------------------
#define LDSS 132       // f32 row pad: (m*132)%64 distinct -> conflict-free
#define SLOTW 20       // dwords per A-frag slot (16 used + pad, 16B aligned)
__global__ void gemm2_kernel(const float* __restrict__ ev,
                             const float* __restrict__ emb,
                             const unsigned short* __restrict__ bhi,
                             const unsigned short* __restrict__ blo,
                             float* __restrict__ outn) {
  __shared__ float ev_s[16 * LDSS];
  __shared__ float emb_s[16 * LDSS];
  __shared__ unsigned int aP[16 * 32 * SLOTW];  // 16 kb-blocks x 32 lanes, 40KB
  int tid = threadIdx.x;
  int e0 = blockIdx.x << 4;
  int b = blockIdx.z;
  const float* evp  = ev  + ((size_t)b * E_ + e0) * H_;
  const float* embp = emb + ((size_t)b * E_ + e0) * H_;
  for (int t = tid; t < 16 * H_; t += 256) {
    int r = t >> 7, c = t & 127;
    ev_s[r * LDSS + c]  = evp[(size_t)r * H_ + c];
    emb_s[r * LDSS + c] = embp[(size_t)r * H_ + c];
  }
  int lane = tid & 31;
  int w = tid >> 5;          // wave id == i-tile
  int m = lane & 15;
  int i = (w << 4) + m;
  v8f acc = {};

  for (int ch = 0; ch < 32; ++ch) {          // 32 chunks x 512 K = 16384
    int chunk_base = ch << 9;
    __syncthreads();                          // prev consume (or init load) done
    // ---- produce split-P A-fragments for this K-chunk (cooperative) ----
    for (int s = tid; s < 512; s += 256) {    // slot = kb*32 + lane2
      int lane2 = s & 31;
      int kb = s >> 5;
      int m2 = lane2 & 15;
      int base0 = chunk_base + (kb << 5) + ((lane2 & 16) ? 16 : 0);
      float evh = ev_s[m2 * LDSS + (base0 >> 7)];   // h fixed within 32-block
      int j0 = base0 & 127;
      const float* er = emb_s + m2 * LDSS;
      unsigned int hdw[8], ldw[8];
#pragma unroll
      for (int d = 0; d < 8; ++d) {
        int jj = j0 + (d << 1);
        float p0 = evh * er[jj];
        float p1 = evh * er[jj + 1];
        unsigned int b0 = __float_as_uint(p0);
        unsigned int b1 = __float_as_uint(p1);
        unsigned int h0 = b0 & 0xFFFF0000u;
        unsigned int h1 = b1 & 0xFFFF0000u;
        hdw[d] = h1 | (b0 >> 16);
        float r0 = p0 - __uint_as_float(h0);
        float r1 = p1 - __uint_as_float(h1);
        ldw[d] = (__float_as_uint(r1) & 0xFFFF0000u) | (__float_as_uint(r0) >> 16);
      }
      unsigned int* dst = aP + s * SLOTW;
      *(uint4*)(dst)      = make_uint4(hdw[0], hdw[1], hdw[2], hdw[3]);
      *(uint4*)(dst + 4)  = make_uint4(hdw[4], hdw[5], hdw[6], hdw[7]);
      *(uint4*)(dst + 8)  = make_uint4(ldw[0], ldw[1], ldw[2], ldw[3]);
      *(uint4*)(dst + 12) = make_uint4(ldw[4], ldw[5], ldw[6], ldw[7]);
    }
    __syncthreads();
    // ---- consume: 3 WMMAs per 32-K block, B frags coalesced from global ----
#pragma unroll 2
    for (int kb = 0; kb < 16; ++kb) {
      const unsigned int* ap = aP + (kb * 32 + lane) * SLOTW;
      Frag fa, fl;
      fa.q[0] = *(const uint4*)(ap);
      fa.q[1] = *(const uint4*)(ap + 4);
      fl.q[0] = *(const uint4*)(ap + 8);
      fl.q[1] = *(const uint4*)(ap + 12);
      size_t bo = (((size_t)w * 512 + (size_t)((ch << 4) + kb)) * 32 + lane) << 4;
      v16bf bh = *(const v16bf*)(bhi + bo);
      v16bf bl = *(const v16bf*)(blo + bo);
      acc = wmma_bf16(fa.v, bh, acc);
      acc = wmma_bf16(fa.v, bl, acc);
      acc = wmma_bf16(fl.v, bh, acc);
    }
  }
  float* C = outn + ((size_t)b * E_ + e0 + ((lane >> 4) << 3)) * H_ + i;
#pragma unroll
  for (int r = 0; r < 8; ++r) C[(size_t)r * H_] = acc[r];
}

// -------- GEMM3: out = (edge2node @ new_n2e + node_state) / norm -----------
// exact f32 WMMA; M=1000 (guarded to 1008), K=2000, N=128
__global__ void gemm3_kernel(const float* __restrict__ e2n,
                             const float* __restrict__ nn,
                             const float* __restrict__ ns,
                             const float* __restrict__ norm,
                             float* __restrict__ out) {
  int lane = threadIdx.x & 31;
  int m = lane & 15;
  int koff = (lane >> 4) << 1;
  int n0 = blockIdx.x << 4;
  int i0 = blockIdx.y << 4;
  int b = blockIdx.z;
  int arow = n0 + m;
  if (arow >= N_) arow = N_ - 1;  // clamped rows feed only discarded outputs
  const float* A  = e2n + ((size_t)b * N_ + arow) * E_ + koff;
  const float* Bp = nn + (size_t)b * E_ * H_ + (size_t)koff * H_ + i0 + m;
  v8f acc = {};
#pragma unroll 4
  for (int k = 0; k < E_; k += 4) {
    float2 av = *(const float2*)(A + k);
    v2f a = {av.x, av.y};
    v2f bb = {Bp[(size_t)k * H_], Bp[(size_t)(k + 1) * H_]};
    acc = wmma4(a, bb, acc);
  }
  int col = i0 + m;
  int rbase = n0 + ((lane >> 4) << 3);
#pragma unroll
  for (int r = 0; r < 8; ++r) {
    int row = rbase + r;
    if (row < N_) {
      float nv = norm[b * N_ + row];
      size_t idx = ((size_t)b * N_ + row) * H_ + col;
      out[idx] = (acc[r] + ns[idx]) / nv;
    }
  }
}

extern "C" void kernel_launch(void* const* d_in, const int* in_sizes, int n_in,
                              void* d_out, int out_size, void* d_ws, size_t ws_size,
                              hipStream_t stream) {
  (void)in_sizes; (void)n_in; (void)out_size; (void)ws_size;
  const float* node_state = (const float*)d_in[0];
  const float* edge_vec   = (const float*)d_in[1];
  const float* node2edge  = (const float*)d_in[2];
  const float* edge2node  = (const float*)d_in[3];
  const float* edge_net   = (const float*)d_in[4];

  const size_t NSW = (size_t)8 * 512 * 32 * 16;       // swizzled bf16 elems (2M)
  const size_t BEH = (size_t)B_ * E_ * H_;
  unsigned short* bhi = (unsigned short*)d_ws;        // 4 MB
  unsigned short* blo = bhi + NSW;                    // 4 MB
  float* emb  = (float*)(blo + NSW);                  // 2 MB
  float* nn   = emb + BEH;                            // 2 MB
  float* norm = nn + BEH;                             // 8 KB
  float* out  = (float*)d_out;

  en_split_kernel<<<dim3(131072 / 256), dim3(256), 0, stream>>>(edge_net, bhi, blo);
  norm_kernel<<<dim3(B_ * N_), dim3(256), 0, stream>>>(edge2node, norm);
  gemm1_kernel<<<dim3(E_ / 16, H_ / 16, B_), dim3(32), 0, stream>>>(node2edge, node_state, emb);
  gemm2_kernel<<<dim3(E_ / 16, 1, B_), dim3(256), 0, stream>>>(edge_vec, emb, bhi, blo, nn);
  gemm3_kernel<<<dim3((N_ + 15) / 16, H_ / 16, B_), dim3(32), 0, stream>>>(edge2node, nn, node_state, norm, out);
}
